// GCNII_lyc_67087389164132
// MI455X (gfx1250) — matrix operand order repaired and checked by reading, not compiled, not run on hardware
//
#include <hip/hip_runtime.h>
#include <hip/hip_bf16.h>

// ---------------------------------------------------------------------------
// GCNII forward for MI455X (gfx1250): bf16 WMMA for adj@h with the shared h
// operand staged into LDS via async global->LDS copies (double buffered,
// ASYNCcnt), theta folded into W_eff, fused per-layer kernel.
// ---------------------------------------------------------------------------

typedef __attribute__((ext_vector_type(16))) __bf16 v16bf;
typedef __attribute__((ext_vector_type(8)))  __bf16 v8bf;
typedef __attribute__((ext_vector_type(4)))  __bf16 v4bf;
typedef __attribute__((ext_vector_type(8)))  float  v8f;
typedef __attribute__((ext_vector_type(4)))  float  v4f;

#define NN      8192     // N nodes
#define NFEAT   512
#define NHID    100
#define HP      128      // padded hidden dim
#define NLAYERS 16

__device__ __forceinline__ v16bf cat16(v8bf lo, v8bf hi) {
  return __builtin_shufflevector(lo, hi, 0,1,2,3,4,5,6,7,8,9,10,11,12,13,14,15);
}
// A-operand (16x32 bf16): per lane 16 contiguous elements
__device__ __forceinline__ v16bf ldA(const __bf16* p) {
  v8bf lo = *(const v8bf*)p;
  v8bf hi = *(const v8bf*)(p + 8);
  return cat16(lo, hi);
}
// B-operand (32x16 bf16): per lane 16 contiguous elements
__device__ __forceinline__ v16bf ldB(const __bf16* p) {
  v8bf lo = *(const v8bf*)p;
  v8bf hi = *(const v8bf*)(p + 8);
  return cat16(lo, hi);
}
__device__ __forceinline__ v8f wmma_bf16(v16bf a, v16bf b, v8f c) {
  return __builtin_amdgcn_wmma_f32_16x16x32_bf16(false, a, false, b,
                                                 (short)0, c, false, false);
}

// ---------- prep kernels ----------

__global__ void adj2bf_kernel(const float* __restrict__ adj,
                              __bf16* __restrict__ out, long n4) {
  long i = blockIdx.x * (long)blockDim.x + threadIdx.x;
  long stride = gridDim.x * (long)blockDim.x;
  for (; i < n4; i += stride) {
    v4f a = ((const v4f*)adj)[i];
    v4bf b;
    #pragma unroll
    for (int j = 0; j < 4; ++j) b[j] = (__bf16)a[j];
    ((v4bf*)out)[i] = b;
  }
}

// W_eff[l] = theta_l * w_l + (1-theta_l) * I, zero-padded to 128x128,
// stored TRANSPOSED: wT[l][n][k] = W_eff[k][n]
__global__ void weff_kernel(const float* __restrict__ conv_ws,
                            __bf16* __restrict__ wT) {
  int t = blockIdx.x * blockDim.x + threadIdx.x;   // 16*128*128 threads
  int k = t & (HP - 1);
  int n = (t >> 7) & (HP - 1);
  int l = t >> 14;
  float theta = __logf(0.5f / (float)(l + 1) + 1.0f);
  float v = 0.f;
  if (k < NHID && n < NHID) {
    v = theta * conv_ws[(size_t)l * NHID * NHID + k * NHID + n];
    if (k == n) v += (1.0f - theta);
  }
  wT[(size_t)l * HP * HP + n * HP + k] = (__bf16)v;
}

// h0 = relu(x @ fc_w + b); f32 row-major (pitch 128, zero pad) + transposed
// bf16 hT0[n][m] (rows n>=100 zero)
__global__ void h0_kernel(const float* __restrict__ x,
                          const float* __restrict__ fcw,
                          const float* __restrict__ fcb,
                          float* __restrict__ h0f,
                          __bf16* __restrict__ hT0) {
  int t = blockIdx.x * blockDim.x + threadIdx.x;   // 8192*128 threads
  int n = t & (HP - 1);
  int m = t >> 7;
  float v = 0.f;
  if (n < NHID) {
    float s = fcb[n];
    const float* xr = x + (size_t)m * NFEAT;
    #pragma unroll 4
    for (int k = 0; k < NFEAT; ++k) s += xr[k] * fcw[k * NHID + n];
    v = s > 0.f ? s : 0.f;
  }
  h0f[(size_t)m * HP + n] = v;
  hT0[(size_t)n * NN + m] = (__bf16)v;
}

// ---------- fused GCNII layer ----------
// grid: 128 blocks x 128 threads (4 waves); wave owns 16 output rows.
// B operand (hT) staged per-WG into LDS with async copies, double buffered.
__global__ __launch_bounds__(128)
void gcn_layer_kernel(const __bf16* __restrict__ adjb,   // 8192x8192 bf16
                      const __bf16* __restrict__ hTin,   // 128x8192 bf16
                      const float*  __restrict__ h0f,    // 8192x128 f32
                      const __bf16* __restrict__ wT,     // 128x128 bf16 (W_eff^T)
                      __bf16* __restrict__ hTout) {      // 128x8192 bf16
  constexpr int KC   = 128;            // K-chunk staged per step
  constexpr int BP   = 136;            // padded LDS pitch (4-bank rotation)
  constexpr int BUFE = 112 * BP;       // elements per staging buffer
  constexpr int NCH  = NN / KC;        // 64 chunks
  __shared__ __bf16 smem[2 * BUFE];    // ~59.5 KB

  const int tid   = threadIdx.x;
  const int lane  = tid & 31;
  const int wave  = tid >> 5;
  const int lhalf = lane >> 4;
  const int lmod  = lane & 15;
  const int mrow  = blockIdx.x * 64 + wave * 16;

  // stage hT[0:112, c*128 : c*128+128] into smem buffer `buf` (async)
  auto prefetch = [&](int c, int buf) {
    const __bf16* gbase = hTin + c * KC;
    uint32_t lbase = (uint32_t)(size_t)(smem + buf * BUFE);
    #pragma unroll
    for (int i = 0; i < 14; ++i) {         // 1792 16B segments / 128 threads
      int seg = i * 128 + tid;
      int n   = seg >> 4;
      int ks  = (seg & 15) * 8;
      const __bf16* g = gbase + (size_t)n * NN + ks;
      uint32_t l = lbase + (uint32_t)(n * BP + ks) * 2;
      asm volatile("global_load_async_to_lds_b128 %0, %1, off"
                   :: "v"(l), "v"(g) : "memory");
    }
  };

  v8f acc[7];
  #pragma unroll
  for (int nt = 0; nt < 7; ++nt)
    #pragma unroll
    for (int r = 0; r < 8; ++r) acc[nt][r] = 0.f;

  const __bf16* aBase = adjb + (size_t)(mrow + lmod) * NN + lhalf * 16;

  // ---- GEMM 1: hi = adj @ h, B from LDS (shared across 4 waves) ----
  prefetch(0, 0);
  for (int c = 0; c < NCH; ++c) {
    if (c + 1 < NCH) {
      prefetch(c + 1, (c + 1) & 1);
      asm volatile("s_wait_asynccnt 14" ::: "memory");  // chunk c landed
    } else {
      asm volatile("s_wait_asynccnt 0" ::: "memory");
    }
    __syncthreads();                                    // visible to all waves
    const __bf16* bBuf = smem + (c & 1) * BUFE;
    #pragma unroll
    for (int kk = 0; kk < KC; kk += 32) {
      v16bf a = ldA(aBase + c * KC + kk);
      #pragma unroll
      for (int nt = 0; nt < 7; ++nt) {
        v16bf b = ldB(bBuf + (nt * 16 + lmod) * BP + kk + lhalf * 16);
        acc[nt] = wmma_bf16(a, b, acc[nt]);
      }
    }
    __syncthreads();                 // everyone done reading before overwrite
  }

  // ---- support = 0.9*hi + 0.1*h0 -> LDS (reuse staging memory) ----
  __bf16* sw = smem + wave * 16 * BP;      // per-wave 16x136 region
  for (int i = lane; i < 16 * 16; i += 32) // zero pad cols 112..127
    sw[(i >> 4) * BP + 112 + (i & 15)] = (__bf16)0.f;
  #pragma unroll
  for (int nt = 0; nt < 7; ++nt) {
    #pragma unroll
    for (int r = 0; r < 8; ++r) {
      int lm = r + 8 * lhalf;
      int n  = nt * 16 + lmod;
      float v = 0.9f * acc[nt][r]
              + 0.1f * h0f[(size_t)(mrow + lm) * HP + n];
      sw[lm * BP + n] = (__bf16)v;
    }
  }
  // same-wave LDS ops are in-order; each wave reads only its own region

  // ---- GEMM 2: h_next = relu(support @ W_eff)  (theta folded in) ----
  v8f acc2[7];
  #pragma unroll
  for (int nt = 0; nt < 7; ++nt)
    #pragma unroll
    for (int r = 0; r < 8; ++r) acc2[nt][r] = 0.f;

  #pragma unroll
  for (int k0 = 0; k0 < HP; k0 += 32) {
    v16bf a = ldA(sw + lmod * BP + k0 + lhalf * 16);
    #pragma unroll
    for (int nt = 0; nt < 7; ++nt) {
      v16bf b = ldB(wT + (nt * 16 + lmod) * HP + k0 + lhalf * 16);
      acc2[nt] = wmma_bf16(a, b, acc2[nt]);
    }
  }
  #pragma unroll
  for (int nt = 0; nt < 7; ++nt) {
    #pragma unroll
    for (int r = 0; r < 8; ++r) {
      float v = acc2[nt][r];
      v = v > 0.f ? v : 0.f;
      int m = mrow + r + 8 * lhalf;
      int n = nt * 16 + lmod;
      hTout[(size_t)n * NN + m] = (__bf16)v;
    }
  }
}

// ---------- epilogue: out = concat([x, h]) ----------
__global__ void out_kernel(const float* __restrict__ x,
                           const __bf16* __restrict__ hTf,
                           float* __restrict__ out) {
  int t = blockIdx.x * blockDim.x + threadIdx.x;   // 8192*612
  if (t >= NN * (NFEAT + NHID)) return;
  int j = t % (NFEAT + NHID);
  int m = t / (NFEAT + NHID);
  out[t] = (j < NFEAT) ? x[(size_t)m * NFEAT + j]
                       : (float)hTf[(size_t)(j - NFEAT) * NN + m];
}

extern "C" void kernel_launch(void* const* d_in, const int* in_sizes, int n_in,
                              void* d_out, int out_size, void* d_ws, size_t ws_size,
                              hipStream_t stream) {
  const float* x       = (const float*)d_in[0];
  const float* adj     = (const float*)d_in[1];
  const float* fc_w    = (const float*)d_in[2];
  const float* fc_b    = (const float*)d_in[3];
  const float* conv_ws = (const float*)d_in[4];

  char* ws = (char*)d_ws;
  size_t off = 0;
  __bf16* adjbf = (__bf16*)(ws + off); off += (size_t)NN * NN * 2;          // 128 MB
  float*  h0f   = (float*) (ws + off); off += (size_t)NN * HP * 4;          // 4 MB
  __bf16* wT    = (__bf16*)(ws + off); off += (size_t)NLAYERS * HP * HP * 2;// 512 KB
  __bf16* hT0   = (__bf16*)(ws + off); off += (size_t)HP * NN * 2;          // 2 MB
  __bf16* hT1   = (__bf16*)(ws + off); off += (size_t)HP * NN * 2;          // 2 MB

  // prep
  adj2bf_kernel<<<8192, 256, 0, stream>>>(adj, adjbf, (long)NN * NN / 4);
  weff_kernel<<<(NLAYERS * HP * HP) / 256, 256, 0, stream>>>(conv_ws, wT);
  h0_kernel<<<(NN * HP) / 256, 256, 0, stream>>>(x, fc_w, fc_b, h0f, hT0);
  hipMemsetAsync(hT1, 0, (size_t)HP * NN * 2, stream);   // pad rows of ping buffer

  // 16 fused layers, ping-pong transposed h buffers
  __bf16* hin = hT0;
  __bf16* hout = hT1;
  for (int l = 0; l < NLAYERS; ++l) {
    gcn_layer_kernel<<<NN / 64, 128, 0, stream>>>(
        adjbf, hin, h0f, wT + (size_t)l * HP * HP, hout);
    __bf16* t = hin; hin = hout; hout = t;
  }

  // concat([x, h_final])
  int total = NN * (NFEAT + NHID);
  out_kernel<<<(total + 255) / 256, 256, 0, stream>>>(x, hin, (float*)d_out);
}
